// DecoderLayer_65807488910062
// MI455X (gfx1250) — compile-verified, matrix operand order checked by reading, not run on hardware
//
#include <hip/hip_runtime.h>
#include <hip/hip_bf16.h>
#include <stdint.h>

// ---------------------------------------------------------------------------
// Types
// ---------------------------------------------------------------------------
typedef __attribute__((ext_vector_type(16))) __bf16 v16bf;
typedef __attribute__((ext_vector_type(8)))  __bf16 v8bf;
typedef __attribute__((ext_vector_type(8)))  float  v8f;
typedef __attribute__((ext_vector_type(4)))  unsigned int u32x4;
typedef __attribute__((ext_vector_type(8)))  int i32x8;
typedef __attribute__((ext_vector_type(4)))  int i32x4;

#if defined(__has_builtin)
#if __has_builtin(__builtin_amdgcn_tensor_load_to_lds) && \
    __has_builtin(__builtin_amdgcn_s_wait_tensorcnt)
#define USE_TDM 1
#endif
#endif

__device__ __forceinline__ uint16_t f32_to_bf16_bits(float f) {
  union { float f; uint32_t u; } v; v.f = f;
  uint32_t u = v.u;
  if ((u & 0x7FFFFFFFu) > 0x7F800000u) return (uint16_t)((u >> 16) | 0x0040u); // NaN
  uint32_t r = u + 0x7FFFu + ((u >> 16) & 1u);   // round-to-nearest-even
  return (uint16_t)(r >> 16);
}

// ---------------------------------------------------------------------------
// f32 -> bf16 conversion (grid-stride)
// ---------------------------------------------------------------------------
__global__ void k_to_bf16(const float* __restrict__ in, uint16_t* __restrict__ out, long n) {
  long i = (long)blockIdx.x * blockDim.x + threadIdx.x;
  long stride = (long)gridDim.x * blockDim.x;
  for (; i < n; i += stride) out[i] = f32_to_bf16_bits(in[i]);
}

// ---------------------------------------------------------------------------
// TDM helper: issue 2D tile DMA (global -> LDS) via Tensor Data Mover.
//   tile0 elements contiguous (dim0), tile1 rows of stride `stride_el`.
//   data_size = 2 bytes.  LDS rows padded: 128B data + 16B pad = 144B stride.
// D# packing per CDNA5 ISA ch. 8 (groups 0/1; groups 2/3 zero for 2D).
// This toolchain's builtin takes 6 args:
//   (u32x4 g0, i32x8 g1, i32x4 g2, i32x4 g3, i32x8 extra, i32 cpol)
// ---------------------------------------------------------------------------
#ifdef USE_TDM
__device__ __forceinline__ void tdm_load_2d(uint64_t gaddr_bytes, uint32_t lds_addr,
                                            int tile0, int tile1, long stride_el) {
  u32x4 g0;
  g0[0] = 1u;                                                // count=1, user mode
  g0[1] = lds_addr;                                          // [63:32] lds_addr
  g0[2] = (uint32_t)(gaddr_bytes & 0xFFFFFFFFull);           // [95:64]
  g0[3] = (uint32_t)((gaddr_bytes >> 32) & 0x01FFFFFFull)    // [120:96]
        | 0x80000000u;                                       // type=2 ("image")
  const uint32_t d0 = 0x7FFFFFFFu, d1 = 0x7FFFFFFFu;         // huge dims: no OOB clip
  const uint64_t s0 = (uint64_t)stride_el;
  i32x8 g1;
  g1[0] = (int)((1u << 16)          // data_size = 1 -> 2 bytes
              | (1u << 20)          // pad_enable
              | (4u << 22)          // pad_interval: 32 DWORDs (=128B) of data
              | (3u << 25));        // pad_amount: 4 DWORDs (=16B)
  g1[1] = (int)((d0 & 0xFFFFu) << 16);                       // tensor_dim0 lo16
  g1[2] = (int)((d0 >> 16) | ((d1 & 0xFFFFu) << 16));        // dim0 hi / dim1 lo
  g1[3] = (int)((d1 >> 16) | ((uint32_t)tile0 << 16));       // dim1 hi / tile_dim0
  g1[4] = (int)((uint32_t)tile1 & 0xFFFFu);                  // tile_dim1 (tile_dim2=0)
  g1[5] = (int)(uint32_t)(s0 & 0xFFFFFFFFull);               // tensor_dim0_stride lo
  g1[6] = (int)(uint32_t)((s0 >> 32) & 0xFFFFull);           // stride hi (dim1_stride=0)
  g1[7] = 0;
  i32x4 z4; z4[0] = 0; z4[1] = 0; z4[2] = 0; z4[3] = 0;
  i32x8 z8; z8[0] = 0; z8[1] = 0; z8[2] = 0; z8[3] = 0;
            z8[4] = 0; z8[5] = 0; z8[6] = 0; z8[7] = 0;
  __builtin_amdgcn_tensor_load_to_lds(g0, g1, z4, z4, z8, 0);
}
__device__ __forceinline__ uint32_t lds_addr_of(const void* p) {
  // generic LDS pointers carry the wave-relative LDS byte address in [31:0]
  return (uint32_t)(uintptr_t)p;
}
#endif

// ---------------------------------------------------------------------------
// Generic batched bf16 WMMA GEMM:  C = A(MxK) * B(KxN) (+bias) [+relu]
//   A row-major lda; TRANSB=0: B row-major ldb; TRANSB=1: element (k,n) at
//   B[n*ldb + k].  Batch offset = (z/batchH)*sXo + (z%batchH)*sXi.
//   Writes f32 (Cf) and/or bf16 (Cb), same ldc/strides.
// Requires M%128==0, N%64==0, K%64==0 (true for all uses here).
// Block 256 thr = 8 waves (4x2), wave tile 32x32, BK=64 -> 8 WMMA / barrier.
// ---------------------------------------------------------------------------
#define BM 128
#define BN 64
#define BKT 64
#define LROW 72   // LDS row stride in elements (144B: 16B-aligned, bank-spread)

__device__ __forceinline__ v16bf load_frag(const uint16_t* p, int half) {
  // per-lane K runs per ISA 7.12.2: [8*half,+8) and [16+8*half,+8)
  v8bf lo = *(const v8bf*)(p + half * 8);
  v8bf hi = *(const v8bf*)(p + 16 + half * 8);
  return __builtin_shufflevector(lo, hi, 0,1,2,3,4,5,6,7,8,9,10,11,12,13,14,15);
}

template <int TRANSB>
__global__ __launch_bounds__(256)
void k_gemm_bf16(const uint16_t* __restrict__ A, int lda, long sAo, long sAi,
                 const uint16_t* __restrict__ Bm, int ldb, long sBo, long sBi,
                 const float* __restrict__ bias,
                 float* __restrict__ Cf, uint16_t* __restrict__ Cb,
                 int ldc, long sCo, long sCi,
                 int M, int N, int K, int batchH, int relu) {
  const int bz = blockIdx.z;
  const int zo = bz / batchH, zi = bz % batchH;
  A  += zo * sAo + zi * sAi;
  Bm += zo * sBo + zi * sBi;
  const long coff = (long)zo * sCo + (long)zi * sCi;

  __shared__ uint16_t As[2][BM][LROW];
  __shared__ uint16_t Bs[2][BN][LROW];

  const int t    = threadIdx.x;
  const int lane = t & 31;
  const int wave = t >> 5;
  const int half = lane >> 4;
  const int l16  = lane & 15;
  const int wm   = wave >> 1;      // 0..3
  const int wn   = wave & 1;       // 0..1
  const int m0   = blockIdx.y * BM;
  const int n0   = blockIdx.x * BN;
  const int nk   = K / BKT;

  v8f acc[2][2] = {};

#ifdef USE_TDM
  // ---------------- TDM double-buffered pipeline ----------------
  const uint64_t abase = (uint64_t)(uintptr_t)(A + (long)m0 * lda);
  const uint64_t bbase = TRANSB ? (uint64_t)(uintptr_t)(Bm + (long)n0 * ldb) : 0ull;

  // stage tile 0
  if (t < 32) {
    tdm_load_2d(abase, lds_addr_of(&As[0][0][0]), BKT, BM, lda);
    if (TRANSB) tdm_load_2d(bbase, lds_addr_of(&Bs[0][0][0]), BKT, BN, ldb);
  }
  if (!TRANSB) {
#pragma unroll
    for (int c = 0; c < 2; ++c) {
      int chunk = t + c * 256;           // 0..511
      int kk = chunk >> 3;               // 0..63
      int nc = (chunk & 7) << 3;         // 0..56
      uint4 d = *(const uint4*)(Bm + (long)kk * ldb + (n0 + nc));
      const uint16_t* dp = (const uint16_t*)&d;
#pragma unroll
      for (int i = 0; i < 8; ++i) Bs[0][nc + i][kk] = dp[i];
    }
  }
  if (t < 32) __builtin_amdgcn_s_wait_tensorcnt(0);
  __syncthreads();

  for (int kt = 0; kt < nk; ++kt) {
    const int cur = kt & 1, nxt = cur ^ 1;
    const int k0n = (kt + 1) * BKT;
    uint4 breg[2];
    if (kt + 1 < nk) {
      if (t < 32) {
        tdm_load_2d(abase + (uint64_t)k0n * 2u, lds_addr_of(&As[nxt][0][0]), BKT, BM, lda);
        if (TRANSB)
          tdm_load_2d(bbase + (uint64_t)k0n * 2u, lds_addr_of(&Bs[nxt][0][0]), BKT, BN, ldb);
      }
      if (!TRANSB) {
#pragma unroll
        for (int c = 0; c < 2; ++c) {
          int chunk = t + c * 256;
          int kk = chunk >> 3;
          int nc = (chunk & 7) << 3;
          breg[c] = *(const uint4*)(Bm + (long)(k0n + kk) * ldb + (n0 + nc));
        }
      }
    }
    // ---- compute: 2 K-chunks x (2x2) WMMA ----
#pragma unroll
    for (int c = 0; c < 2; ++c) {
      v16bf afrag[2], bfrag[2];
#pragma unroll
      for (int mi = 0; mi < 2; ++mi)
        afrag[mi] = load_frag(&As[cur][wm * 32 + mi * 16 + l16][c * 32], half);
#pragma unroll
      for (int nj = 0; nj < 2; ++nj)
        bfrag[nj] = load_frag(&Bs[cur][wn * 32 + nj * 16 + l16][c * 32], half);
#pragma unroll
      for (int mi = 0; mi < 2; ++mi)
#pragma unroll
        for (int nj = 0; nj < 2; ++nj)
          acc[mi][nj] = __builtin_amdgcn_wmma_f32_16x16x32_bf16(
              false, afrag[mi], false, bfrag[nj], (short)0, acc[mi][nj], false, false);
    }
    if (kt + 1 < nk && !TRANSB) {
#pragma unroll
      for (int c = 0; c < 2; ++c) {
        int chunk = t + c * 256;
        int kk = chunk >> 3;
        int nc = (chunk & 7) << 3;
        const uint16_t* dp = (const uint16_t*)&breg[c];
#pragma unroll
        for (int i = 0; i < 8; ++i) Bs[nxt][nc + i][kk] = dp[i];
      }
    }
    if (t < 32) __builtin_amdgcn_s_wait_tensorcnt(0);
    __syncthreads();
  }
#else
  // ---------------- fallback: synchronous staging ----------------
  for (int kt = 0; kt < nk; ++kt) {
    const int k0 = kt * BKT;
#pragma unroll
    for (int c = 0; c < 4; ++c) {
      int chunk = t + c * 256;           // 0..1023 : A tile 128x64
      int row = chunk >> 3;
      int kc  = (chunk & 7) << 3;
      uint4 d = *(const uint4*)(A + (long)(m0 + row) * lda + (k0 + kc));
      *(uint4*)&As[0][row][kc] = d;
    }
#pragma unroll
    for (int c = 0; c < 2; ++c) {
      int chunk = t + c * 256;
      if (TRANSB) {
        int n = chunk >> 3, kc = (chunk & 7) << 3;
        uint4 d = *(const uint4*)(Bm + (long)(n0 + n) * ldb + (k0 + kc));
        *(uint4*)&Bs[0][n][kc] = d;
      } else {
        int kk = chunk >> 3, nc = (chunk & 7) << 3;
        uint4 d = *(const uint4*)(Bm + (long)(k0 + kk) * ldb + (n0 + nc));
        const uint16_t* dp = (const uint16_t*)&d;
#pragma unroll
        for (int i = 0; i < 8; ++i) Bs[0][nc + i][kk] = dp[i];
      }
    }
    __syncthreads();
#pragma unroll
    for (int c = 0; c < 2; ++c) {
      v16bf afrag[2], bfrag[2];
#pragma unroll
      for (int mi = 0; mi < 2; ++mi)
        afrag[mi] = load_frag(&As[0][wm * 32 + mi * 16 + l16][c * 32], half);
#pragma unroll
      for (int nj = 0; nj < 2; ++nj)
        bfrag[nj] = load_frag(&Bs[0][wn * 32 + nj * 16 + l16][c * 32], half);
#pragma unroll
      for (int mi = 0; mi < 2; ++mi)
#pragma unroll
        for (int nj = 0; nj < 2; ++nj)
          acc[mi][nj] = __builtin_amdgcn_wmma_f32_16x16x32_bf16(
              false, afrag[mi], false, bfrag[nj], (short)0, acc[mi][nj], false, false);
    }
    __syncthreads();
  }
#endif

  // ---- epilogue: C layout (ISA 7.12.2): VGPR e -> M = e + 8*half, N = l16 ----
#pragma unroll
  for (int mi = 0; mi < 2; ++mi) {
#pragma unroll
    for (int nj = 0; nj < 2; ++nj) {
      const int col = n0 + wn * 32 + nj * 16 + l16;
      const float bv = bias ? bias[col] : 0.0f;
#pragma unroll
      for (int e = 0; e < 8; ++e) {
        const int row = m0 + wm * 32 + mi * 16 + half * 8 + e;
        float v = acc[mi][nj][e] + bv;
        if (relu) v = v > 0.0f ? v : 0.0f;
        const long idx = coff + (long)row * ldc + col;
        if (Cf) Cf[idx] = v;
        if (Cb) Cb[idx] = f32_to_bf16_bits(v);
      }
    }
  }
}

// ---------------------------------------------------------------------------
// Block reductions (wave32 shfl + cross-wave LDS)
// ---------------------------------------------------------------------------
__device__ __forceinline__ float wave_sum(float x) {
#pragma unroll
  for (int o = 16; o > 0; o >>= 1) x += __shfl_xor(x, o, 32);
  return x;
}
__device__ __forceinline__ float wave_max(float x) {
#pragma unroll
  for (int o = 16; o > 0; o >>= 1) x = fmaxf(x, __shfl_xor(x, o, 32));
  return x;
}
__device__ __forceinline__ float block_allreduce(float x, float* s, bool ismax) {
  x = ismax ? wave_max(x) : wave_sum(x);
  const int lane = threadIdx.x & 31, w = threadIdx.x >> 5;
  if (lane == 0) s[w] = x;
  __syncthreads();
  float r = s[0];
#pragma unroll
  for (int i = 1; i < 8; ++i) r = ismax ? fmaxf(r, s[i]) : (r + s[i]);
  __syncthreads();
  return r;
}

// ---------------------------------------------------------------------------
// Masked softmax over one attention row (in place on f32 logits) + bf16 copy.
// ---------------------------------------------------------------------------
__global__ __launch_bounds__(256)
void k_softmax_mask(float* __restrict__ logits, uint16_t* __restrict__ probs_bf,
                    const float* __restrict__ mask, int mask_per_q,
                    int S_, int Tc, int Hh, float scale) {
  __shared__ float sred[8];
  const long row = blockIdx.x;
  const int q = (int)(row % S_);
  const int b = (int)(row / ((long)Hh * S_));
  float* lp = logits + row * (long)Tc;
  uint16_t* pb = probs_bf + row * (long)Tc;
  const float* mrow = mask + (mask_per_q ? (long)q * Tc : (long)b * Tc);
  const int t = threadIdx.x;

  float v[4];
  float mx = -3.0e38f;
#pragma unroll
  for (int i = 0; i < 4; ++i) {
    const int c = t + i * 256;
    const float x = lp[c] * scale + mrow[c] * (-1.0e9f);
    v[i] = x;
    mx = fmaxf(mx, x);
  }
  mx = block_allreduce(mx, sred, true);
  float sum = 0.0f;
#pragma unroll
  for (int i = 0; i < 4; ++i) { v[i] = __expf(v[i] - mx); sum += v[i]; }
  sum = block_allreduce(sum, sred, false);
  const float inv = 1.0f / sum;
#pragma unroll
  for (int i = 0; i < 4; ++i) {
    const int c = t + i * 256;
    const float p = v[i] * inv;
    lp[c] = p;
    pb[c] = f32_to_bf16_bits(p);
  }
}

// ---------------------------------------------------------------------------
// out = LayerNorm(a + b) * g + be ; writes f32 and optional bf16 copy.
// ---------------------------------------------------------------------------
__global__ __launch_bounds__(256)
void k_add_ln(const float* __restrict__ a, const float* __restrict__ b,
              const float* __restrict__ g, const float* __restrict__ be,
              float* __restrict__ of, uint16_t* __restrict__ ob,
              int Dd, float eps) {
  __shared__ float sred[8];
  const long row = blockIdx.x;
  const float* ap = a + row * (long)Dd;
  const float* bp = b + row * (long)Dd;
  const int t = threadIdx.x;
  float v[4], s = 0.0f, s2 = 0.0f;
#pragma unroll
  for (int i = 0; i < 4; ++i) {
    const int c = t + i * 256;
    const float x = ap[c] + bp[c];
    v[i] = x; s += x; s2 += x * x;
  }
  s  = block_allreduce(s,  sred, false);
  s2 = block_allreduce(s2, sred, false);
  const float mu  = s / (float)Dd;
  const float var = s2 / (float)Dd - mu * mu;
  const float r   = rsqrtf(var + eps);
#pragma unroll
  for (int i = 0; i < 4; ++i) {
    const int c = t + i * 256;
    const float y = (v[i] - mu) * r * g[c] + be[c];
    of[row * (long)Dd + c] = y;
    if (ob) ob[row * (long)Dd + c] = f32_to_bf16_bits(y);
  }
}

// ---------------------------------------------------------------------------
// Host orchestration
// ---------------------------------------------------------------------------
extern "C" void kernel_launch(void* const* d_in, const int* in_sizes, int n_in,
                              void* d_out, int out_size, void* d_ws, size_t ws_size,
                              hipStream_t stream) {
  (void)in_sizes; (void)n_in; (void)out_size; (void)ws_size;
  const long Bb = 4, S = 1024, T = 1024, D = 1024, H = 16, DFF = 4096, DEPTH = 64;
  const long ND   = Bb * S * D;
  const long NW   = D * D;
  const long NFF  = D * DFF;
  const long NATT = Bb * H * S * S;
  const long NH   = Bb * S * DFF;
  const float eps = 1e-3f;
  const float scale = 0.125f;          // 1/sqrt(64)

  const float* x        = (const float*)d_in[0];
  const float* enc      = (const float*)d_in[1];
  const float* la_mask  = (const float*)d_in[2];
  const float* pad_mask = (const float*)d_in[3];
  const float* m_wq = (const float*)d_in[4];  const float* m_bq = (const float*)d_in[5];
  const float* m_wk = (const float*)d_in[6];  const float* m_bk = (const float*)d_in[7];
  const float* m_wv = (const float*)d_in[8];  const float* m_bv = (const float*)d_in[9];
  const float* m_wo = (const float*)d_in[10]; const float* m_bo = (const float*)d_in[11];
  const float* c_wq = (const float*)d_in[12]; const float* c_bq = (const float*)d_in[13];
  const float* c_wk = (const float*)d_in[14]; const float* c_bk = (const float*)d_in[15];
  const float* c_wv = (const float*)d_in[16]; const float* c_bv = (const float*)d_in[17];
  const float* c_wo = (const float*)d_in[18]; const float* c_bo = (const float*)d_in[19];
  const float* fw1  = (const float*)d_in[20]; const float* fb1  = (const float*)d_in[21];
  const float* fw2  = (const float*)d_in[22]; const float* fb2  = (const float*)d_in[23];
  const float* ln1g = (const float*)d_in[24]; const float* ln1b = (const float*)d_in[25];
  const float* ln2g = (const float*)d_in[26]; const float* ln2b = (const float*)d_in[27];
  const float* ln3g = (const float*)d_in[28]; const float* ln3b = (const float*)d_in[29];

  float* out3   = (float*)d_out;
  float* m_attn = out3 + ND;
  float* c_attn = m_attn + NATT;

  uint8_t* wsb = (uint8_t*)d_ws;
  size_t off = 0;
  auto abf = [&](long n) { uint16_t* p = (uint16_t*)(wsb + off);
                           off += (((size_t)n * 2) + 255) & ~(size_t)255; return p; };
  auto af  = [&](long n) { float* p = (float*)(wsb + off);
                           off += (((size_t)n * 4) + 255) & ~(size_t)255; return p; };

  uint16_t* x_bf   = abf(ND);   uint16_t* enc_bf = abf(ND);
  uint16_t* wmq    = abf(NW);   uint16_t* wmk    = abf(NW);
  uint16_t* wmv    = abf(NW);   uint16_t* wmo    = abf(NW);
  uint16_t* wcq    = abf(NW);   uint16_t* wck    = abf(NW);
  uint16_t* wcv    = abf(NW);   uint16_t* wco    = abf(NW);
  uint16_t* w1b    = abf(NFF);  uint16_t* w2b    = abf(NFF);
  uint16_t* q_bf   = abf(ND);   uint16_t* k_bf   = abf(ND);   uint16_t* v_bf  = abf(ND);
  uint16_t* ctx_bf = abf(ND);
  uint16_t* out1_bf = abf(ND);
  uint16_t* cq_bf  = abf(ND);   uint16_t* ck_bf  = abf(ND);   uint16_t* cv_bf = abf(ND);
  uint16_t* cctx_bf = abf(ND);
  uint16_t* out2_bf = abf(ND);
  uint16_t* ffnh_bf = abf(NH);
  uint16_t* attnS_bf = abf(NATT);
  uint16_t* attnC_bf = abf(NATT);
  float* m_out_f = af(ND);  float* out1_f = af(ND);
  float* c_out_f = af(ND);  float* out2_f = af(ND);
  float* ffn2_f  = af(ND);

  auto cvt = [&](const float* in, uint16_t* o, long n) {
    k_to_bf16<<<dim3(4096), dim3(256), 0, stream>>>(in, o, n);
  };
  auto gemm = [&](const uint16_t* A, int lda, long sAo, long sAi,
                  const uint16_t* Bp, int ldb, long sBo, long sBi, int transB,
                  const float* bias, float* Cf, uint16_t* Cb,
                  int ldc, long sCo, long sCi,
                  int M, int N, int K, int batch, int batchH, int relu) {
    dim3 grid(N / BN, M / BM, batch);
    if (transB)
      k_gemm_bf16<1><<<grid, dim3(256), 0, stream>>>(A, lda, sAo, sAi, Bp, ldb, sBo, sBi,
                                                     bias, Cf, Cb, ldc, sCo, sCi,
                                                     M, N, K, batchH, relu);
    else
      k_gemm_bf16<0><<<grid, dim3(256), 0, stream>>>(A, lda, sAo, sAi, Bp, ldb, sBo, sBi,
                                                     bias, Cf, Cb, ldc, sCo, sCi,
                                                     M, N, K, batchH, relu);
  };

  // ---- 0. precision conversion -------------------------------------------
  cvt(x, x_bf, ND);     cvt(enc, enc_bf, ND);
  cvt(m_wq, wmq, NW);   cvt(m_wk, wmk, NW); cvt(m_wv, wmv, NW); cvt(m_wo, wmo, NW);
  cvt(c_wq, wcq, NW);   cvt(c_wk, wck, NW); cvt(c_wv, wcv, NW); cvt(c_wo, wco, NW);
  cvt(fw1, w1b, NFF);   cvt(fw2, w2b, NFF);

  const int Mrows = (int)(Bb * S);     // 4096
  // ---- 1. masked self-attention ------------------------------------------
  gemm(x_bf, D, 0,0, wmq, D, 0,0, 0, m_bq, nullptr, q_bf, D, 0,0, Mrows, D, D, 1,1, 0);
  gemm(x_bf, D, 0,0, wmk, D, 0,0, 0, m_bk, nullptr, k_bf, D, 0,0, Mrows, D, D, 1,1, 0);
  gemm(x_bf, D, 0,0, wmv, D, 0,0, 0, m_bv, nullptr, v_bf, D, 0,0, Mrows, D, D, 1,1, 0);
  gemm(q_bf, D, S*D, DEPTH, k_bf, D, S*D, DEPTH, 1, nullptr, m_attn, nullptr,
       (int)S, H*S*S, S*S, (int)S, (int)S, (int)DEPTH, (int)(Bb*H), (int)H, 0);
  k_softmax_mask<<<dim3((unsigned)(Bb*H*S)), dim3(256), 0, stream>>>(
      m_attn, attnS_bf, la_mask, 1, (int)S, (int)S, (int)H, scale);
  gemm(attnS_bf, (int)S, H*S*S, S*S, v_bf, D, S*D, DEPTH, 0, nullptr, nullptr, ctx_bf,
       D, S*D, DEPTH, (int)S, (int)DEPTH, (int)S, (int)(Bb*H), (int)H, 0);
  gemm(ctx_bf, D, 0,0, wmo, D, 0,0, 0, m_bo, m_out_f, nullptr, D, 0,0, Mrows, D, D, 1,1, 0);
  k_add_ln<<<dim3((unsigned)(Bb*S)), dim3(256), 0, stream>>>(
      m_out_f, x, ln1g, ln1b, out1_f, out1_bf, (int)D, eps);

  // ---- 2. cross-attention -------------------------------------------------
  gemm(out1_bf, D, 0,0, wcq, D, 0,0, 0, c_bq, nullptr, cq_bf, D, 0,0, Mrows, D, D, 1,1, 0);
  gemm(enc_bf,  D, 0,0, wck, D, 0,0, 0, c_bk, nullptr, ck_bf, D, 0,0, Mrows, D, D, 1,1, 0);
  gemm(enc_bf,  D, 0,0, wcv, D, 0,0, 0, c_bv, nullptr, cv_bf, D, 0,0, Mrows, D, D, 1,1, 0);
  gemm(cq_bf, D, S*D, DEPTH, ck_bf, D, T*D, DEPTH, 1, nullptr, c_attn, nullptr,
       (int)T, H*S*T, S*T, (int)S, (int)T, (int)DEPTH, (int)(Bb*H), (int)H, 0);
  k_softmax_mask<<<dim3((unsigned)(Bb*H*S)), dim3(256), 0, stream>>>(
      c_attn, attnC_bf, pad_mask, 0, (int)S, (int)T, (int)H, scale);
  gemm(attnC_bf, (int)T, H*S*T, S*T, cv_bf, D, T*D, DEPTH, 0, nullptr, nullptr, cctx_bf,
       D, S*D, DEPTH, (int)S, (int)DEPTH, (int)T, (int)(Bb*H), (int)H, 0);
  gemm(cctx_bf, D, 0,0, wco, D, 0,0, 0, c_bo, c_out_f, nullptr, D, 0,0, Mrows, D, D, 1,1, 0);
  k_add_ln<<<dim3((unsigned)(Bb*S)), dim3(256), 0, stream>>>(
      c_out_f, out1_f, ln2g, ln2b, out2_f, out2_bf, (int)D, eps);

  // ---- 3. FFN -------------------------------------------------------------
  gemm(out2_bf, D, 0,0, w1b, (int)DFF, 0,0, 0, fb1, nullptr, ffnh_bf,
       (int)DFF, 0,0, Mrows, (int)DFF, (int)D, 1,1, 1 /*relu*/);
  gemm(ffnh_bf, (int)DFF, 0,0, w2b, D, 0,0, 0, fb2, ffn2_f, nullptr,
       D, 0,0, Mrows, (int)D, (int)DFF, 1,1, 0);
  k_add_ln<<<dim3((unsigned)(Bb*S)), dim3(256), 0, stream>>>(
      ffn2_f, out2_f, ln3g, ln3b, out3, nullptr, (int)D, eps);
}
